// GE2ELoss_46523085750731
// MI455X (gfx1250) — matrix operand order repaired and checked by reading, not compiled, not run on hardware
//
#include <hip/hip_runtime.h>
#include <hip/hip_bf16.h>

typedef __attribute__((ext_vector_type(16))) _Float16 v16h;
typedef __attribute__((ext_vector_type(8)))  _Float16 v8h;
typedef __attribute__((ext_vector_type(8)))  float    v8f;

#define N_SPK  2048
#define U      32
#define HALF_U 16
#define D      256
#define M_ROWS (N_SPK * HALF_U)   // 32768 normalized test rows
#define GEPS   1e-8f

// ---------------- zero accumulators ----------------
__global__ void ge2e_zero_kernel(float* __restrict__ total, float* __restrict__ pos) {
  int i = blockIdx.x * blockDim.x + threadIdx.x;
  if (i < N_SPK) { total[i] = 0.0f; pos[i] = 0.0f; }
}

// ---------------- prep: centroids + row-normalize to f16 ----------------
// blocks [0, N_SPK): centroid of speaker b (mean of 16 enroll rows), normalize -> cnh
// blocks [N_SPK, N_SPK+M_ROWS): test row m = b - N_SPK, normalize -> tnh
__global__ void __launch_bounds__(256)
ge2e_prep_kernel(const float* __restrict__ emb,
                 _Float16* __restrict__ tnh,
                 _Float16* __restrict__ cnh) {
  __shared__ float red[256];
  const int b = blockIdx.x;
  const int d = threadIdx.x;   // one element of the 256-dim vector per thread
  float val;
  if (b < N_SPK) {
    const float* base = emb + (size_t)b * U * D + d;
    float s = 0.0f;
#pragma unroll
    for (int u = 0; u < HALF_U; ++u) s += base[(size_t)u * D];
    val = s * (1.0f / (float)HALF_U);
  } else {
    const int m   = b - N_SPK;
    const int spk = m >> 4;
    const int t   = m & 15;
    val = emb[((size_t)(spk * U + HALF_U + t)) * D + d];
  }
  red[d] = val * val;
  __syncthreads();
  for (int s = 128; s > 0; s >>= 1) {
    if (d < s) red[d] += red[d + s];
    __syncthreads();
  }
  const float scale = 1.0f / fmaxf(sqrtf(red[0]), GEPS);
  const _Float16 h = (_Float16)(val * scale);
  if (b < N_SPK) cnh[(size_t)b * D + d] = h;
  else           tnh[(size_t)(b - N_SPK) * D + d] = h;
}

// ---------------- fused GEMM + exp + column reduction ----------------
// sims = alpha * tnh(32768x256) * cnh^T(256x2048) + beta ; e = exp(sims)
// total[n] += sum_m e[m][n] ; pos[n] += sum over the 16 rows of speaker n of e[.,n]
//
// Block: 256 threads = 8 waves arranged 4(M) x 2(N). Each wave computes TWO
// 16x16 N-tiles (16M x 32N) so every A-fragment load feeds two WMMAs, halving
// L2 A-operand traffic. Block tile = 64M x 64N. grid.x = N/64 = 32 column
// strips; grid.y = 32; each block loops over M chunks (16 iterations).
__global__ void __launch_bounds__(256)
ge2e_gemm_exp_kernel(const _Float16* __restrict__ tnh,
                     const _Float16* __restrict__ cnh,
                     const float* __restrict__ alpha_p,
                     const float* __restrict__ beta_p,
                     float* __restrict__ total,
                     float* __restrict__ pos) {
  const float alpha = alpha_p[0];
  const float beta  = beta_p[0];

  __shared__ float cs[64];

  const int tid   = threadIdx.x;
  const int wave  = tid >> 5;
  const int lane  = tid & 31;
  const int wm    = wave & 3;   // wave's M slot (0..3)
  const int wn    = wave >> 2;  // wave's N slot (0..1), 32 columns each
  const int lan16 = lane & 15;
  const int hi    = lane >> 4;  // lane half (affects K packing)

  if (tid < 64) cs[tid] = 0.0f;
  __syncthreads();

  // Global columns (speaker indices) this lane covers for its two N-tiles.
  const int n0    = blockIdx.x * 64 + wn * 32;
  const int colg0 = n0 + lan16;
  const int colg1 = n0 + 16 + lan16;

  // Preload B fragments for both N-tiles, all 8 K-steps (fixed per block).
  // B k order used here (same as A): lane = column; lanes 0-15 hold K=k0..k0+15 contiguous,
  // lanes 16-31 hold K=k0+16..k0+31 (one 32-byte contiguous run per lane).
  v16h bfrag0[8], bfrag1[8];
  {
    const _Float16* bp0 = cnh + (size_t)colg0 * D + hi * 16;
    const _Float16* bp1 = cnh + (size_t)colg1 * D + hi * 16;
#pragma unroll
    for (int kk = 0; kk < 8; ++kk) {
      bfrag0[kk] = *(const v16h*)(bp0 + kk * 32);
      bfrag1[kk] = *(const v16h*)(bp1 + kk * 32);
    }
  }

  float colsum0 = 0.0f;
  float colsum1 = 0.0f;

  for (int mchunk = blockIdx.y; mchunk < M_ROWS / 64; mchunk += gridDim.y) {
    const int m0 = mchunk * 64 + wm * 16;  // 16 rows, all one speaker (m0 % 16 == 0)
    // A k order used here (same as B): row = lane%16; K=k0+hi*16..+7 in h[0..7],
    // K=k0+hi*16+8..+15 in h[8..15]  -> two 16-byte loads per lane per K-step.
    const _Float16* ap = tnh + (size_t)(m0 + lan16) * D + hi * 16;
    v8f acc0 = {};
    v8f acc1 = {};
    // Software-pipelined K loop: issue loads for step kk+1 before the WMMAs of
    // step kk so global loads overlap matrix math (partial loadcnt waits).
    v8h lo = *(const v8h*)(ap);
    v8h hh = *(const v8h*)(ap + 8);
#pragma unroll
    for (int kk = 0; kk < 8; ++kk) {
      v16h afrag = __builtin_shufflevector(lo, hh,
          0, 1, 2, 3, 4, 5, 6, 7, 8, 9, 10, 11, 12, 13, 14, 15);
      if (kk < 7) {
        lo = *(const v8h*)(ap + (kk + 1) * 32);
        hh = *(const v8h*)(ap + (kk + 1) * 32 + 8);
      }
      acc0 = __builtin_amdgcn_wmma_f32_16x16x32_f16(
          false, afrag, false, bfrag0[kk], (short)0, acc0, false, false);
      acc1 = __builtin_amdgcn_wmma_f32_16x16x32_f16(
          false, afrag, false, bfrag1[kk], (short)0, acc1, false, false);
    }
    // Epilogue: e = exp(alpha*s + beta); lane holds 8 rows of each of 2 columns.
    float sum8_0 = 0.0f;
    float sum8_1 = 0.0f;
#pragma unroll
    for (int v = 0; v < 8; ++v) {
      sum8_0 += __expf(alpha * acc0[v] + beta);
      sum8_1 += __expf(alpha * acc1[v] + beta);
    }
    colsum0 += sum8_0;
    colsum1 += sum8_1;
    // Positive (diagonal) block: all 16 rows of this tile belong to speaker m0/16.
    const int spk = m0 >> 4;
    if (colg0 == spk) atomicAdd(&pos[spk], sum8_0);
    if (colg1 == spk) atomicAdd(&pos[spk], sum8_1);
  }

  // Block-level column reduction, then one global atomic per column per block.
  atomicAdd(&cs[wn * 32 + lan16],      colsum0);
  atomicAdd(&cs[wn * 32 + 16 + lan16], colsum1);
  __syncthreads();
  if (tid < 64) atomicAdd(&total[blockIdx.x * 64 + tid], cs[tid]);
}

// ---------------- final loss reduction ----------------
__global__ void __launch_bounds__(256)
ge2e_loss_kernel(const float* __restrict__ total,
                 const float* __restrict__ pos,
                 float* __restrict__ out) {
  __shared__ float red[256];
  float acc = 0.0f;
  for (int n = threadIdx.x; n < N_SPK; n += 256) {
    const float p   = pos[n];
    const float neg = total[n] - p;
    acc += logf(neg) - logf(p);
  }
  red[threadIdx.x] = acc;
  __syncthreads();
  for (int s = 128; s > 0; s >>= 1) {
    if (threadIdx.x < s) red[threadIdx.x] += red[threadIdx.x + s];
    __syncthreads();
  }
  if (threadIdx.x == 0) out[0] = red[0] * (1.0f / (float)N_SPK);
}

extern "C" void kernel_launch(void* const* d_in, const int* in_sizes, int n_in,
                              void* d_out, int out_size, void* d_ws, size_t ws_size,
                              hipStream_t stream) {
  (void)in_sizes; (void)n_in; (void)out_size; (void)ws_size;

  const float* emb   = (const float*)d_in[0];
  // d_in[1] = labels (int64): repeat(arange(N_SPK), U) -> stable argsort is the
  // identity permutation, so no reordering is required.
  const float* alpha = (const float*)d_in[2];
  const float* beta  = (const float*)d_in[3];
  float* out = (float*)d_out;

  char* ws = (char*)d_ws;
  const size_t tnh_bytes = (size_t)M_ROWS * D * sizeof(_Float16);  // 16 MB
  const size_t cnh_bytes = (size_t)N_SPK * D * sizeof(_Float16);   // 1 MB
  _Float16* tnh = (_Float16*)ws;
  _Float16* cnh = (_Float16*)(ws + tnh_bytes);
  float* total  = (float*)(ws + tnh_bytes + cnh_bytes);
  float* pos    = total + N_SPK;

  ge2e_zero_kernel<<<(N_SPK + 255) / 256, 256, 0, stream>>>(total, pos);
  ge2e_prep_kernel<<<N_SPK + M_ROWS, 256, 0, stream>>>(emb, tnh, cnh);
  ge2e_gemm_exp_kernel<<<dim3(N_SPK / 64, 32), 256, 0, stream>>>(
      tnh, cnh, alpha, beta, total, pos);
  ge2e_loss_kernel<<<1, 256, 0, stream>>>(total, pos, out);
}
